// Net_34359738709
// MI455X (gfx1250) — compile-verified, matrix operand order checked by reading, not run on hardware
//
#include <hip/hip_runtime.h>
#include <hip/hip_bf16.h>

// ---- problem constants ------------------------------------------------------
#define NIN    32768
#define NH     100
#define NOUT   11
#define TSTEPS 50
#define BATCH  64
#define MROWS  (TSTEPS * BATCH)     // 3200
#define NPAD   112                  // 7 tiles of 16 cols (NH=100 zero-padded)
#define NTILES 7
#define KSPLIT 4                    // K-chunks across blocks  -> 800 blocks
#define KBLK   (NIN / KSPLIT)       // 8192 K per block
#define KWAVE  (KBLK / 8)           // 1024 K per wave (8 waves split-K in block)

typedef __bf16          bf16x16 __attribute__((ext_vector_type(16)));
typedef unsigned short  u16x16  __attribute__((ext_vector_type(16)));
typedef float           f32x8   __attribute__((ext_vector_type(8)));

// round-to-nearest-even f32 -> bf16 (bit trick; fine for this data range)
static __device__ __forceinline__ unsigned short f2bf(float f) {
    unsigned int u = __builtin_bit_cast(unsigned int, f);
    u += 0x7FFFu + ((u >> 16) & 1u);
    return (unsigned short)(u >> 16);
}

// ---- kernel 1: W1 f32 -> bf16, zero-padded to 112 rows (L2-resident, 7.3MB) -
__global__ __launch_bounds__(256) void w1_to_bf16(
    const float* __restrict__ W1, unsigned short* __restrict__ W1bf)
{
    size_t i = (size_t)blockIdx.x * 256 + threadIdx.x;   // over 112*32768
    int h = (int)(i >> 15);
    int k = (int)(i & 32767);
    W1bf[i] = (h < NH) ? f2bf(W1[(size_t)h * NIN + k]) : (unsigned short)0;
}

// ---- kernel 2: partial GEMM  part[kc][m][h] = x[m, kc-chunk] @ W1[h, kc-chunk]
// block = 16 M-rows x all 112 cols x one K/4 chunk; 8 waves split the chunk.
// bf16 WMMA with f32 accumulate; deterministic LDS tree reduction per block.
__global__ __launch_bounds__(256) void snn_cur1_gemm(
    const float* __restrict__ x,
    const unsigned short* __restrict__ W1bf,
    float* __restrict__ part)
{
    __shared__ float red[8 * 16 * NPAD];          // 57 KB: per-wave partials

    const int tid   = threadIdx.x;
    const int wave  = tid >> 5;
    const int lane  = tid & 31;
    const int mrow  = lane & 15;                  // M (and N) index within tile
    const int khalf = (lane >> 4) * 8;            // ISA 16-bit A/B lane striping
    const int mtile = (int)blockIdx.x >> 2;       // 0..199
    const int kc    = (int)blockIdx.x & (KSPLIT - 1);
    const int mbase = mtile * 16;

    f32x8 acc[NTILES];
    #pragma unroll
    for (int t = 0; t < NTILES; ++t)
        #pragma unroll
        for (int i = 0; i < 8; ++i) acc[t][i] = 0.0f;

    const size_t kwbase = (size_t)kc * KBLK + (size_t)wave * KWAVE + khalf;
    const float* arow = x + (size_t)(mbase + mrow) * NIN + kwbase;
    const unsigned short* brow0 = W1bf + (size_t)mrow * NIN + kwbase;

    for (int ks = 0; ks < KWAVE; ks += 32) {
        // unconditional far-ahead prefetch of the A stream (global_prefetch_b8;
        // speculative per ISA: translation faults silently dropped, no LOADcnt).
        // 32 lanes cover the whole 16x128B A tile 4 K-steps ahead.
        __builtin_prefetch(arow + ks + 128, 0, 1);

        // A fragment: 16 f32 per lane (k = kb..kb+7, kb+16..kb+23), cvt to bf16
        const float4 a0 = *(const float4*)(arow + ks);
        const float4 a1 = *(const float4*)(arow + ks + 4);
        const float4 a2 = *(const float4*)(arow + ks + 16);
        const float4 a3 = *(const float4*)(arow + ks + 20);
        u16x16 au;
        au[0]  = f2bf(a0.x); au[1]  = f2bf(a0.y); au[2]  = f2bf(a0.z); au[3]  = f2bf(a0.w);
        au[4]  = f2bf(a1.x); au[5]  = f2bf(a1.y); au[6]  = f2bf(a1.z); au[7]  = f2bf(a1.w);
        au[8]  = f2bf(a2.x); au[9]  = f2bf(a2.y); au[10] = f2bf(a2.z); au[11] = f2bf(a2.w);
        au[12] = f2bf(a3.x); au[13] = f2bf(a3.y); au[14] = f2bf(a3.z); au[15] = f2bf(a3.w);
        bf16x16 afrag = __builtin_bit_cast(bf16x16, au);

        // B fragment loader: column n = t*16+mrow is a row of the bf16 W1 copy
        auto loadB = [&](int t) -> bf16x16 {
            const unsigned short* brow = brow0 + (size_t)(t * 16) * NIN + ks;
            union { u16x16 v; uint4 q[2]; } bb;
            bb.q[0] = *(const uint4*)(brow);        // k = kb .. kb+7
            bb.q[1] = *(const uint4*)(brow + 16);   // k = kb+16 .. kb+23
            return __builtin_bit_cast(bf16x16, bb.v);
        };

#if defined(__HIP_DEVICE_COMPILE__)
        {
            // tile 0: plain; tiles 1..6 share the same A registers -> reuse_a
            bf16x16 bfrag = loadB(0);
            acc[0] = __builtin_amdgcn_wmma_f32_16x16x32_bf16(
                false, afrag, false, bfrag, (short)0, acc[0], false, false);
            #pragma unroll
            for (int t = 1; t < NTILES; ++t) {
                bfrag = loadB(t);
                acc[t] = __builtin_amdgcn_wmma_f32_16x16x32_bf16(
                    false, afrag, false, bfrag, (short)0, acc[t], true, false);
            }
        }
#else
        (void)loadB;
#endif
    }

    // spill per-wave partials: VGPR i of C/D holds M = i + 8*(lane>=16), N = lane&15
    {
        const int col  = lane & 15;
        const int rbas = (lane >> 4) * 8;
        #pragma unroll
        for (int t = 0; t < NTILES; ++t)
            #pragma unroll
            for (int i = 0; i < 8; ++i)
                red[wave * (16 * NPAD) + (rbas + i) * NPAD + t * 16 + col] = acc[t][i];
    }
    __syncthreads();

    // deterministic tree-sum of the 8 intra-block split-K partials, store h<100
    for (int idx = tid; idx < 16 * NPAD; idx += 256) {
        float s = 0.0f;
        #pragma unroll
        for (int w = 0; w < 8; ++w) s += red[w * (16 * NPAD) + idx];
        const int row = idx / NPAD, h = idx % NPAD;
        if (h < NH)
            part[((size_t)kc * MROWS + mbase + row) * NH + h] = s;
    }
}

// ---- kernel 2b: deterministic cross-block K reduction + bias ----------------
__global__ __launch_bounds__(256) void cur1_reduce(
    const float* __restrict__ part, const float* __restrict__ b1,
    float* __restrict__ cur1)
{
    const int i = (int)blockIdx.x * 256 + threadIdx.x;   // over 3200*100
    if (i >= MROWS * NH) return;
    float s = b1[i % NH];
    #pragma unroll
    for (int c = 0; c < KSPLIT; ++c) s += part[(size_t)c * MROWS * NH + i];
    cur1[i] = s;
}

// ---- kernel 3: sequential SNN recurrence (one block per batch row) ----------
__global__ __launch_bounds__(128) void snn_recur(
    const float* __restrict__ cur1,
    const float* __restrict__ beta1,
    const float* __restrict__ V,
    const float* __restrict__ Vb,
    const float* __restrict__ W2,
    const float* __restrict__ b2,
    float* __restrict__ out)
{
    __shared__ float spk[128];
    const int b   = blockIdx.x;
    const int tid = threadIdx.x;
    const bool h_on = tid < NH;
    const bool o_on = tid < NOUT;

    float bet = 0.f, vb = 0.f;
    if (h_on) {
        bet = beta1[tid];
        bet = bet < 0.f ? 0.f : (bet > 1.f ? 1.f : bet);   // snntorch clamp
        vb  = Vb[tid];
    }
    float mem1 = 0.f, mem2 = 0.f;
    spk[tid] = 0.f;
    __syncthreads();

    for (int t = 0; t < TSTEPS; ++t) {
        // recurrent drive from PREVIOUS spikes
        float rec = 0.f;
        if (h_on) {
            const float* vr = V + tid * NH;
            for (int k = 0; k < NH; ++k) rec += spk[k] * vr[k];
        }
        __syncthreads();                       // all reads of old spk done
        if (h_on) {
            const float c1 = cur1[((size_t)t * BATCH + b) * NH + tid];
            const float reset1 = (mem1 - 1.0f > 0.f) ? 1.0f : 0.f;  // pre-update mem
            mem1 = bet * mem1 + c1 + rec + vb - reset1;             // THRESH = 1
            spk[tid] = (mem1 - 1.0f > 0.f) ? 1.0f : 0.f;
        }
        __syncthreads();                       // new spk1 visible
        if (o_on) {
            const float* wr = W2 + tid * NH;
            float c2 = b2[tid];
            for (int k = 0; k < NH; ++k) c2 += spk[k] * wr[k];
            const float reset2 = (mem2 - 1.0f > 0.f) ? 1.0f : 0.f;
            mem2 = 0.9753f * mem2 + c2 - reset2;
            const float s2 = (mem2 - 1.0f > 0.f) ? 1.0f : 0.f;
            const size_t o = ((size_t)t * BATCH + b) * NOUT + tid;
            out[o] = s2;                                  // spk_rec
            out[(size_t)TSTEPS * BATCH * NOUT + o] = mem2; // mem_rec
        }
    }
}

// ---- launch -----------------------------------------------------------------
extern "C" void kernel_launch(void* const* d_in, const int* in_sizes, int n_in,
                              void* d_out, int out_size, void* d_ws, size_t ws_size,
                              hipStream_t stream) {
    const float* x     = (const float*)d_in[0];
    const float* W1    = (const float*)d_in[1];
    const float* b1    = (const float*)d_in[2];
    const float* beta1 = (const float*)d_in[3];
    const float* V     = (const float*)d_in[4];
    const float* Vb    = (const float*)d_in[5];
    const float* W2    = (const float*)d_in[6];
    const float* b2    = (const float*)d_in[7];

    // workspace layout (all 16B aligned):
    //   cur1 : 3200*100 f32            = 1,280,000 B
    //   W1bf : 112*32768 bf16          = 7,340,032 B
    //   part : 4*3200*100 f32          = 5,120,000 B
    float*          cur1 = (float*)d_ws;
    unsigned short* W1bf = (unsigned short*)((char*)d_ws + 1280000);
    float*          partb = (float*)((char*)d_ws + 1280000 + 7340032);

    w1_to_bf16   <<<(NPAD * NIN) / 256,      256, 0, stream>>>(W1, W1bf);
    snn_cur1_gemm<<<(MROWS / 16) * KSPLIT,   256, 0, stream>>>(x, W1bf, partb);
    cur1_reduce  <<<(MROWS * NH + 255) / 256,256, 0, stream>>>(partb, b1, cur1);
    snn_recur    <<<BATCH,                   128, 0, stream>>>(cur1, beta1, V, Vb, W2, b2,
                                                               (float*)d_out);
    (void)in_sizes; (void)n_in; (void)out_size; (void)ws_size;
}
